// ReferModel_28613072126639
// MI455X (gfx1250) — compile-verified, matrix operand order checked by reading, not run on hardware
//
#include <hip/hip_runtime.h>
#include <hip/hip_bf16.h>

typedef __attribute__((ext_vector_type(16))) _Float16 v16h;
typedef __attribute__((ext_vector_type(8)))  float    v8f;
typedef int i32x4 __attribute__((vector_size(16)));   // matches builtin prototype

#define EPSF 1e-5f
#define BB   16
#define TT   256
#define DD   128
#define HH   128
#define LL   4
#define G4   (4*HH)   // 512

__device__ __forceinline__ float sigmoidf_(float x) { return 1.0f / (1.0f + expf(-x)); }

// K-index inside a 16x32 f16 WMMA operand for element e=2r+p, half hi (lane>>4).
// VGPR0..3 : K = hi*8 + 2r + p ; VGPR4..7 : K = 16 + hi*8 + 2(r-4) + p
__device__ __forceinline__ int wmma_kidx(int r, int hi) {
    return (r < 4) ? (hi * 8 + 2 * r) : (16 + hi * 8 + 2 * (r - 4));
}

__device__ __forceinline__ v16h load_frag(const _Float16* p, int hi) {
    v16h f{};
#pragma unroll
    for (int r = 0; r < 8; ++r) {
        int kk = wmma_kidx(r, hi);
        f[2 * r]     = p[kk];
        f[2 * r + 1] = p[kk + 1];
    }
    return f;
}

// ---- gfx1250 async global->LDS copy (ASYNCcnt path), with safe fallbacks ----
__device__ __forceinline__ void async_copy_b128(const float* g, float* l) {
#if __has_builtin(__builtin_amdgcn_global_load_async_to_lds_b128)
    __builtin_amdgcn_global_load_async_to_lds_b128(
        (__attribute__((address_space(1))) i32x4*)g,
        (__attribute__((address_space(3))) i32x4*)l,
        0, 0);
#else
    *(float4*)l = *(const float4*)g;
#endif
}
__device__ __forceinline__ void wait_async0() {
#if __has_builtin(__builtin_amdgcn_s_wait_asynccnt)
    __builtin_amdgcn_s_wait_asynccnt(0);
#elif __has_builtin(__builtin_amdgcn_global_load_async_to_lds_b128)
    asm volatile("s_wait_asynccnt 0" ::: "memory");
#endif
}

// ---------------------------------------------------------------------------
// Weight prep: fp32 -> f16 copies of Wih/Whh/proj weights, bias sum.
// ---------------------------------------------------------------------------
__global__ void prep_kernel(const float* Wih, const float* Whh,
                            const float* bih, const float* bhh,
                            const float* phw, const float* pxw,
                            _Float16* WihH, _Float16* WhhH, float* bsum,
                            _Float16* phwH, _Float16* pxwH) {
    int i0 = blockIdx.x * blockDim.x + threadIdx.x;
    int stride = gridDim.x * blockDim.x;
    for (int i = i0; i < LL * G4 * HH; i += stride) {
        WihH[i] = (_Float16)Wih[i];
        WhhH[i] = (_Float16)Whh[i];
    }
    for (int i = i0; i < HH * HH; i += stride) {
        phwH[i] = (_Float16)phw[i];
        pxwH[i] = (_Float16)pxw[i];
    }
    for (int i = i0; i < LL * G4; i += stride) bsum[i] = bih[i] + bhh[i];
}

// ---------------------------------------------------------------------------
// Per-feature stats over (B,T): torch std with ddof=1, stored as std+EPS.
// grid = 128 (one per feature d), block = 256
// ---------------------------------------------------------------------------
__global__ void featstats_kernel(const float* x, float* fm, float* fs) {
    __shared__ float rs[256], rq[256];
    int d = blockIdx.x, tid = threadIdx.x;
    float s = 0.f, q = 0.f;
    for (int i = tid; i < BB * TT; i += 256) {
        float v = x[(size_t)i * DD + d];
        s += v; q += v * v;
    }
    rs[tid] = s; rq[tid] = q; __syncthreads();
    for (int o = 128; o > 0; o >>= 1) {
        if (tid < o) { rs[tid] += rs[tid + o]; rq[tid] += rq[tid + o]; }
        __syncthreads();
    }
    if (tid == 0) {
        float n = (float)(BB * TT);
        float mean = rs[0] / n;
        float var = (rq[0] - n * mean * mean) / (n - 1.0f);
        fm[d] = mean;
        fs[d] = sqrtf(var) + EPSF;
    }
}

// ---------------------------------------------------------------------------
// InstanceNorm1d over T per (b,c) on [B,T,C]; optional feature pre-scaling
// (fm/fs) applied first; optional f32 and f16 outputs. grid=B*C, block=T=256
// ---------------------------------------------------------------------------
__global__ void instnorm_kernel(const float* X, const float* fm, const float* fs,
                                float* Y, _Float16* Yh, int C) {
    __shared__ float rs[256], rq[256];
    int bc = blockIdx.x;
    int b = bc / C, c = bc % C;
    int t = threadIdx.x;
    size_t idx = ((size_t)b * TT + t) * C + c;
    float v = X[idx];
    if (fm) v = (v - fm[c]) / fs[c];
    rs[t] = v; rq[t] = v * v; __syncthreads();
    for (int o = 128; o > 0; o >>= 1) {
        if (t < o) { rs[t] += rs[t + o]; rq[t] += rq[t + o]; }
        __syncthreads();
    }
    float mean = rs[0] / (float)TT;
    float var = rq[0] / (float)TT - mean * mean;   // biased, as nn.InstanceNorm1d
    float y = (v - mean) * rsqrtf(var + EPSF);
    if (Y)  Y[idx] = y;
    if (Yh) Yh[idx] = (_Float16)y;
}

// ---------------------------------------------------------------------------
// WMMA GEMM: C[M,N] = A[M,K](f16,row-major) @ Bw[N,K](f16,row-major)^T + bias.
// Each wave computes 1 M-tile x 4 N-tiles: A fragment reused across 4 WMMAs.
// 8 waves/block. K multiple of 32, N multiple of 64.
// ---------------------------------------------------------------------------
__global__ void gemm_wmma_kernel(const _Float16* __restrict__ A,
                                 const _Float16* __restrict__ Bw,
                                 const float* __restrict__ bias,
                                 float* __restrict__ C, int M, int N, int K) {
    int lane = threadIdx.x & 31;
    int ng   = N >> 6;                               // groups of 4 n-tiles
    int wid  = blockIdx.x * 8 + (threadIdx.x >> 5);
    if (wid >= (M >> 4) * ng) return;                // wave-uniform exit
    int tm = wid / ng, tg = wid % ng;
    int m0 = tm << 4, n0 = tg << 6;
    int row = lane & 15, hi = lane >> 4;

    v8f acc[4] = {};
    for (int k0 = 0; k0 < K; k0 += 32) {
        v16h a = load_frag(A + (size_t)(m0 + row) * K + k0, hi);
#pragma unroll
        for (int j = 0; j < 4; ++j) {
            v16h b = load_frag(Bw + (size_t)(n0 + 16 * j + row) * K + k0, hi);
            acc[j] = __builtin_amdgcn_wmma_f32_16x16x32_f16(false, a, false, b,
                                                            (short)0, acc[j],
                                                            false, false);
        }
    }
#pragma unroll
    for (int j = 0; j < 4; ++j)
#pragma unroll
        for (int r = 0; r < 8; ++r) {
            int m = m0 + r + 8 * hi;
            int n = n0 + 16 * j + row;
            C[(size_t)m * N + n] = acc[j][r] + (bias ? bias[n] : 0.0f);
        }
}

// ---------------------------------------------------------------------------
// One LSTM layer recurrence. Gx = X@Wih.T + b precomputed ([B,T,4H]).
// Single workgroup (16 waves, 512 threads). h,c persistent in LDS.
// The 32KB Gx slice for step t is DMA'd into LDS with async loads while the
// WMMA chain of step t runs; the gate epilogue accumulates in place into that
// buffer, so staging buffer == gate buffer (LDS total 44KB).
// ---------------------------------------------------------------------------
__global__ void __launch_bounds__(512)
lstm_layer_kernel(const float* __restrict__ Gx,
                  const _Float16* __restrict__ Whh,   // [4H,H] f16
                  float* __restrict__ Hout,           // [B,T,H] f32
                  _Float16* __restrict__ HoutH) {     // [B,T,H] f16 (next layer)
    __shared__ _Float16 hsh[BB * HH];     // 4 KB  h (f16, WMMA A operand)
    __shared__ float    csh[BB * HH];     // 8 KB  cell state
    __shared__ float    gsh[BB * G4];     // 32 KB async-staged Gx slice -> gates

    const int tid = threadIdx.x;
    const int lane = tid & 31, wave = tid >> 5;
    const int row = lane & 15, hi = lane >> 4;

    for (int i = tid; i < BB * HH; i += 512) { hsh[i] = (_Float16)0.f; csh[i] = 0.f; }
    for (int i = tid; i < G4; i += 512)                 // warm Whh (global_prefetch)
        __builtin_prefetch(Whh + (size_t)i * HH, 0, 3);
    __syncthreads();

    // stage Gx slice for t=0 (each thread: 4 x b128 async copies)
#pragma unroll
    for (int c = 0; c < 4; ++c) {
        int chunk = tid + 512 * c;                      // 0..2047 chunks of 4 floats
        int b = chunk >> 7, n = (chunk & 127) << 2;
        async_copy_b128(Gx + ((size_t)b * TT + 0) * G4 + n, gsh + b * G4 + n);
    }

    const int n0a = (wave * 2) << 4;                    // this wave's 2 n-tiles
    for (int t = 0; t < TT; ++t) {
        v8f acc0 = {}, acc1 = {};
        for (int k0 = 0; k0 < HH; k0 += 32) {           // overlaps in-flight async
            v16h a  = load_frag(hsh + row * HH + k0, hi);
            v16h b0 = load_frag(Whh + (size_t)(n0a + row) * HH + k0, hi);
            v16h b1 = load_frag(Whh + (size_t)(n0a + 16 + row) * HH + k0, hi);
            acc0 = __builtin_amdgcn_wmma_f32_16x16x32_f16(false, a, false, b0,
                                                          (short)0, acc0, false, false);
            acc1 = __builtin_amdgcn_wmma_f32_16x16x32_f16(false, a, false, b1,
                                                          (short)0, acc1, false, false);
        }
        wait_async0();        // own async b128s landed in LDS
        __syncthreads();      // all waves: slice visible, hsh reads done
#pragma unroll
        for (int r = 0; r < 8; ++r) {                   // gates = Gx + h@Whh.T (in place)
            int m = r + 8 * hi;
            gsh[m * G4 + n0a + row]      += acc0[r];
            gsh[m * G4 + n0a + 16 + row] += acc1[r];
        }
        __syncthreads();      // gates complete

        for (int e = tid; e < BB * HH; e += 512) {
            int b = e >> 7, j = e & (HH - 1);
            float gi = gsh[b * G4 + j];
            float gf = gsh[b * G4 + HH + j];
            float gg = gsh[b * G4 + 2 * HH + j];
            float go = gsh[b * G4 + 3 * HH + j];
            float c = sigmoidf_(gf) * csh[e] + sigmoidf_(gi) * tanhf(gg);
            float h = sigmoidf_(go) * tanhf(c);
            csh[e] = c;
            hsh[e] = (_Float16)h;
            size_t oidx = ((size_t)b * TT + t) * HH + j;
            Hout[oidx]  = h;
            HoutH[oidx] = (_Float16)h;
        }
        __syncthreads();      // gsh reads + h/c updates done

        if (t + 1 < TT) {     // kick off DMA of next slice; overlaps next WMMA
#pragma unroll
            for (int c = 0; c < 4; ++c) {
                int chunk = tid + 512 * c;
                int b = chunk >> 7, n = (chunk & 127) << 2;
                async_copy_b128(Gx + ((size_t)b * TT + (t + 1)) * G4 + n,
                                gsh + b * G4 + n);
            }
        }
    }
}

// ---------------------------------------------------------------------------
// Attention + softmax + context + FC. Broadcast over D collapses:
// alpha[b,t,d] = softmax_T(s)[b,t] / D ;  context = sum_t ph[b,t,:]*p[t].
// grid = B, block = 256 (= T)
// ---------------------------------------------------------------------------
__global__ void attention_kernel(const float* __restrict__ ph,
                                 const float* __restrict__ px,
                                 const float* __restrict__ attn_w,
                                 const float* __restrict__ fc_w,
                                 const float* __restrict__ fc_b,
                                 float* __restrict__ out,      // [B,10]
                                 float* __restrict__ amap) {   // [B,T,D]
    __shared__ float prob[256];
    __shared__ float red[256];
    __shared__ float ctx[HH];
    int b = blockIdx.x, t = threadIdx.x;
    const float* phb = ph + (size_t)b * TT * HH;
    const float* pxb = px + (size_t)b * TT * HH;

    float s = 0.f;
    for (int h = 0; h < HH; ++h)
        s += tanhf(phb[t * HH + h] + pxb[t * HH + h]) * attn_w[h];

    red[t] = s; __syncthreads();
    for (int o = 128; o > 0; o >>= 1) {
        if (t < o) red[t] = fmaxf(red[t], red[t + o]);
        __syncthreads();
    }
    float mx = red[0]; __syncthreads();
    float e = expf(s - mx);
    red[t] = e; __syncthreads();
    for (int o = 128; o > 0; o >>= 1) {
        if (t < o) red[t] += red[t + o];
        __syncthreads();
    }
    float p = e / red[0];
    prob[t] = p; __syncthreads();

    float pd = p / (float)DD;
    float* am = amap + ((size_t)b * TT + t) * DD;
    for (int d = 0; d < DD; ++d) am[d] = pd;

    if (t < HH) {
        float c = 0.f;
        for (int tt = 0; tt < TT; ++tt) c += phb[tt * HH + t] * prob[tt];
        ctx[t] = c;
    }
    __syncthreads();
    if (t < 10) {
        float o = fc_b[t];
        for (int h = 0; h < HH; ++h) o += ctx[h] * fc_w[t * HH + h];
        out[b * 10 + t] = o;
    }
}

// ---------------------------------------------------------------------------
extern "C" void kernel_launch(void* const* d_in, const int* in_sizes, int n_in,
                              void* d_out, int out_size, void* d_ws, size_t ws_size,
                              hipStream_t stream) {
    const float* x      = (const float*)d_in[0];
    const float* Whh    = (const float*)d_in[2];
    const float* bih    = (const float*)d_in[3];
    const float* bhh    = (const float*)d_in[4];
    const float* phb_w  = (const float*)d_in[6];
    const float* pxb_w  = (const float*)d_in[8];
    const float* attn_w = (const float*)d_in[9];
    const float* fc_w   = (const float*)d_in[10];
    const float* fc_b   = (const float*)d_in[11];
    float* out_final = (float*)d_out;            // [B,10] then [B,T,D] flat

    char* w = (char*)d_ws;
    auto alloc = [&](size_t bytes) {
        void* p = (void*)w;
        w += (bytes + 255) & ~(size_t)255;
        return p;
    };
    const size_t NBT = (size_t)BB * TT;                   // 4096
    float*    xn    = (float*)   alloc(NBT * DD * 4);
    _Float16* xnh   = (_Float16*)alloc(NBT * DD * 2);
    float*    Gx    = (float*)   alloc(NBT * G4 * 4);
    float*    hbuf  = (float*)   alloc(NBT * HH * 4);
    _Float16* hbufh = (_Float16*)alloc(NBT * HH * 2);
    _Float16* lnh   = (_Float16*)alloc(NBT * HH * 2);
    float*    phbuf = (float*)   alloc(NBT * HH * 4);
    float*    pxraw = (float*)   alloc(NBT * HH * 4);
    float*    pxbuf = (float*)   alloc(NBT * HH * 4);
    _Float16* WihH  = (_Float16*)alloc((size_t)LL * G4 * HH * 2);
    _Float16* WhhH  = (_Float16*)alloc((size_t)LL * G4 * HH * 2);
    float*    bsum  = (float*)   alloc((size_t)LL * G4 * 4);
    _Float16* phwH  = (_Float16*)alloc((size_t)HH * HH * 2);
    _Float16* pxwH  = (_Float16*)alloc((size_t)HH * HH * 2);
    float*    fm    = (float*)   alloc(DD * 4);
    float*    fs    = (float*)   alloc(DD * 4);
    (void)ws_size; (void)in_sizes; (void)n_in; (void)out_size;

    // 1) weight conversion
    prep_kernel<<<256, 256, 0, stream>>>((const float*)d_in[1], Whh, bih, bhh,
                                         (const float*)d_in[5], (const float*)d_in[7],
                                         WihH, WhhH, bsum, phwH, pxwH);
    // 2) detached feature stats + instance norm of x
    featstats_kernel<<<DD, 256, 0, stream>>>(x, fm, fs);
    instnorm_kernel<<<BB * DD, 256, 0, stream>>>(x, fm, fs, xn, xnh, DD);

    // 3) LSTM stack: hoisted input GEMM + persistent-workgroup recurrence
    for (int l = 0; l < LL; ++l) {
        const _Float16* inh = (l == 0) ? xnh : hbufh;
        int blocks = (int)(((NBT / 16) * (G4 / 64) + 7) / 8);
        gemm_wmma_kernel<<<blocks, 256, 0, stream>>>(
            inh, WihH + (size_t)l * G4 * HH, bsum + (size_t)l * G4,
            Gx, (int)NBT, G4, HH);
        lstm_layer_kernel<<<1, 512, 0, stream>>>(Gx, WhhH + (size_t)l * G4 * HH,
                                                 hbuf, hbufh);
    }

    // 4) instnorm(lstm_out) -> f16 for proj_h GEMM
    instnorm_kernel<<<BB * HH, 256, 0, stream>>>(hbuf, nullptr, nullptr,
                                                 nullptr, lnh, HH);
    // 5) ph = norm(lstm)@proj_h_w.T + b ; px = instnorm(xn@proj_x_w.T + b)
    int blocks2 = (int)(((NBT / 16) * (HH / 64) + 7) / 8);
    gemm_wmma_kernel<<<blocks2, 256, 0, stream>>>(lnh, phwH, phb_w, phbuf,
                                                  (int)NBT, HH, HH);
    gemm_wmma_kernel<<<blocks2, 256, 0, stream>>>(xnh, pxwH, pxb_w, pxraw,
                                                  (int)NBT, HH, HH);
    instnorm_kernel<<<BB * HH, 256, 0, stream>>>(pxraw, nullptr, nullptr,
                                                 pxbuf, nullptr, HH);

    // 6) attention + softmax + context + fc  (outputs: [B,10] ++ [B,T,D])
    attention_kernel<<<BB, 256, 0, stream>>>(phbuf, pxbuf, attn_w, fc_w, fc_b,
                                             out_final, out_final + BB * 10);
}
